// RowSelfAttention_59863254172650
// MI455X (gfx1250) — compile-verified, matrix operand order checked by reading, not run on hardware
//
#include <hip/hip_runtime.h>
#include <hip/hip_bf16.h>
#include <math.h>

// ---------------- problem constants ----------------
constexpr int R = 128;       // rows (tied axis)
constexpr int C = 256;       // columns (attention axis)
constexpr int E = 768;       // embed dim
constexpr int H = 12;        // heads
constexpr int D = 64;        // head dim
constexpr int MTOK = R * C;  // 32768 tokens
constexpr int KQ = R * D;    // 8192 : contraction length for tied logits
constexpr int NQKV = 3 * E;  // 2304 : fused QKV output width

typedef __attribute__((ext_vector_type(16))) __bf16 v16bf;
typedef __attribute__((ext_vector_type(8)))  float  v8f;
typedef __attribute__((ext_vector_type(4)))  float  v4f;
typedef __attribute__((ext_vector_type(4)))  __bf16 v4bf;

// ---------------- WMMA fragment helpers ----------------
// 16-bit A/B operand layout (ISA 7.12.2): lane = (M or N) & 15, half = lane>>4.
// Element e of the 16-wide bf16 vector maps to K = (e<8?0:16) + 8*half + (e&7).
// Per lane: two contiguous 8-element chunks at k0+8*half and k0+8*half+16.

__device__ __forceinline__ v16bf load_frag(const __bf16* __restrict__ p) {
  v16bf f;
#pragma unroll
  for (int e = 0; e < 8; ++e) {
    f[e]     = p[e];
    f[e + 8] = p[e + 16];
  }
  return f;
}

__device__ __forceinline__ v8f wmma_bf16(v16bf a, v16bf b, v8f c) {
  return __builtin_amdgcn_wmma_f32_16x16x32_bf16(
      /*neg_a=*/false, a, /*neg_b=*/false, b,
      /*c_mod=*/(short)0, c, /*reuse_a=*/false, /*reuse_b=*/false);
}

// ---------------- 4x2 register-blocked NT-GEMM core, double-buffered ----------------
// Wave computes a 64x32 tile of Cmat = A(MxK, row-major) * B^T(NxK, row-major).
// Fragments for chunk k0+32 are loaded before the 8 WMMAs of chunk k0 issue,
// so each wait covers a full iteration of WMMA latency.
__device__ __forceinline__ void gemm_core_4x2(const __bf16* __restrict__ A, size_t lda,
                                              const __bf16* __restrict__ B, size_t ldb,
                                              int m0, int n0, int K,
                                              v8f acc[4][2]) {
  const int lane   = threadIdx.x & 31;
  const int lane16 = lane & 15;
  const int half   = lane >> 4;

  const __bf16* ar[4];
  const __bf16* br[2];
#pragma unroll
  for (int mt = 0; mt < 4; ++mt)
    ar[mt] = A + (size_t)(m0 + mt * 16 + lane16) * lda + half * 8;
#pragma unroll
  for (int nt = 0; nt < 2; ++nt)
    br[nt] = B + (size_t)(n0 + nt * 16 + lane16) * ldb + half * 8;

  v16bf a[4], b[2];
#pragma unroll
  for (int mt = 0; mt < 4; ++mt) a[mt] = load_frag(ar[mt]);
#pragma unroll
  for (int nt = 0; nt < 2; ++nt) b[nt] = load_frag(br[nt]);

  for (int k0 = 0; k0 < K - 32; k0 += 32) {
    const int kn = k0 + 32;
    if (kn + 32 < K) {  // prefetch one chunk past the in-flight loads
#pragma unroll
      for (int mt = 0; mt < 4; ++mt) __builtin_prefetch(ar[mt] + kn + 32, 0, 1);
#pragma unroll
      for (int nt = 0; nt < 2; ++nt) __builtin_prefetch(br[nt] + kn + 32, 0, 1);
    }
    v16bf an[4], bn[2];
#pragma unroll
    for (int mt = 0; mt < 4; ++mt) an[mt] = load_frag(ar[mt] + kn);
#pragma unroll
    for (int nt = 0; nt < 2; ++nt) bn[nt] = load_frag(br[nt] + kn);

#pragma unroll
    for (int mt = 0; mt < 4; ++mt)
#pragma unroll
      for (int nt = 0; nt < 2; ++nt)
        acc[mt][nt] = wmma_bf16(a[mt], b[nt], acc[mt][nt]);

#pragma unroll
    for (int mt = 0; mt < 4; ++mt) a[mt] = an[mt];
#pragma unroll
    for (int nt = 0; nt < 2; ++nt) b[nt] = bn[nt];
  }

  // peeled final chunk
#pragma unroll
  for (int mt = 0; mt < 4; ++mt)
#pragma unroll
    for (int nt = 0; nt < 2; ++nt)
      acc[mt][nt] = wmma_bf16(a[mt], b[nt], acc[mt][nt]);
}

// ---------------- kernel 0a: x -> bf16 (4 elements/thread) ----------------
__global__ void xcvt_kernel(const float* __restrict__ x, __bf16* __restrict__ xb) {
  size_t i = ((size_t)blockIdx.x * blockDim.x + threadIdx.x) * 4;
  v4f v = *(const v4f*)(x + i);
  v4bf o;
#pragma unroll
  for (int e = 0; e < 4; ++e) o[e] = (__bf16)v[e];
  *(v4bf*)(xb + i) = o;
}

// ---------------- kernel 0b: transpose weight to bf16 (N x K) ----------------
__global__ void wtrans_kernel(const float* __restrict__ w, __bf16* __restrict__ wT) {
  int idx = blockIdx.x * blockDim.x + threadIdx.x;   // over E*E outputs
  int n = idx / E;       // output row  (original column)
  int k = idx % E;       // output col  (original row)
  wT[(size_t)n * E + k] = (__bf16)w[(size_t)k * E + n];
}

// ---------------- kernel 1: fused QKV projection with layout scatter ----------------
// y = x @ [wq|wk|wv] + [bq|bk|bv]; per output segment:
//   seg 0 (q): dst q_t[h][c][r*D+d], scaled by D^-0.5/sqrt(R)
//   seg 1 (k): dst k_t[h][c][r*D+d]
//   seg 2 (v): dst v_t[h][r][d][c]  (transposed for context stage)
__global__ void qkv_kernel(const __bf16* __restrict__ xb,
                           const __bf16* __restrict__ wqkvT,
                           const float* __restrict__ bq,
                           const float* __restrict__ bk,
                           const float* __restrict__ bv,
                           __bf16* __restrict__ q_t,
                           __bf16* __restrict__ k_t,
                           __bf16* __restrict__ v_t,
                           float q_scale) {
  const int lane16 = threadIdx.x & 15;
  const int half   = (threadIdx.x >> 4) & 1;
  const int wave   = blockIdx.x * (blockDim.x >> 5) + (threadIdx.x >> 5);
  const int nw_n   = NQKV / 32;              // 72 wave-cols
  const int m0     = (wave / nw_n) * 64;     // 512 wave-rows
  const int n0     = (wave % nw_n) * 32;

  v8f acc[4][2] = {};
  gemm_core_4x2(xb, (size_t)E, wqkvT, (size_t)E, m0, n0, E, acc);

#pragma unroll
  for (int mt = 0; mt < 4; ++mt)
#pragma unroll
    for (int nt = 0; nt < 2; ++nt)
#pragma unroll
      for (int e = 0; e < 8; ++e) {
        int mi = m0 + mt * 16 + e + 8 * half;   // token index = r*C + c
        int ni = n0 + nt * 16 + lane16;         // 0..2303
        int seg = ni / E;                       // 0=q 1=k 2=v
        int f   = ni % E;                       // feature within segment
        int r = mi / C, c = mi % C;
        int h = f / D,  d = f % D;
        if (seg == 0) {
          float val = (acc[mt][nt][e] + bq[f]) * q_scale;
          q_t[((size_t)h * C + c) * (size_t)KQ + (size_t)r * D + d] = (__bf16)val;
        } else if (seg == 1) {
          float val = acc[mt][nt][e] + bk[f];
          k_t[((size_t)h * C + c) * (size_t)KQ + (size_t)r * D + d] = (__bf16)val;
        } else {
          float val = acc[mt][nt][e] + bv[f];
          v_t[(((size_t)h * R + r) * D + d) * C + c] = (__bf16)val;
        }
      }
}

// ---------------- kernel 2: tied logits  logits[h][i][j] = sum_{r,d} q*k ----------------
__global__ void logits_kernel(const __bf16* __restrict__ q_t,
                              const __bf16* __restrict__ k_t,
                              float* __restrict__ logits) {
  const int lane16 = threadIdx.x & 15;
  const int half   = (threadIdx.x >> 4) & 1;
  int wave = blockIdx.x * (blockDim.x >> 5) + (threadIdx.x >> 5);
  const int h  = wave / 32;        // 12 heads, 32 waves/head
  const int t  = wave % 32;
  const int m0 = (t / 8) * 64;     // 4 wave-rows * 64 = 256 (i)
  const int n0 = (t % 8) * 32;     // 8 wave-cols * 32 = 256 (j)

  const __bf16* A = q_t + (size_t)h * C * KQ;
  const __bf16* B = k_t + (size_t)h * C * KQ;

  v8f acc[4][2] = {};
  gemm_core_4x2(A, (size_t)KQ, B, (size_t)KQ, m0, n0, KQ, acc);

#pragma unroll
  for (int mt = 0; mt < 4; ++mt)
#pragma unroll
    for (int nt = 0; nt < 2; ++nt)
#pragma unroll
      for (int e = 0; e < 8; ++e) {
        int i = m0 + mt * 16 + e + 8 * half;
        int j = n0 + nt * 16 + lane16;
        logits[((size_t)h * C + i) * C + j] = acc[mt][nt][e];
      }
}

// ---------------- kernel 3: softmax over j, write bf16 probs ----------------
__global__ void softmax_kernel(const float* __restrict__ logits,
                               __bf16* __restrict__ probs) {
  const int row = blockIdx.x;       // h*C + i  (3072 rows)
  const int j   = threadIdx.x;      // 256 columns
  __shared__ float red[256];

  float v = logits[(size_t)row * C + j];
  red[j] = v;
  __syncthreads();
  for (int s = 128; s > 0; s >>= 1) {
    if (j < s) red[j] = fmaxf(red[j], red[j + s]);
    __syncthreads();
  }
  const float mx = red[0];
  __syncthreads();
  const float ex = expf(v - mx);
  red[j] = ex;
  __syncthreads();
  for (int s = 128; s > 0; s >>= 1) {
    if (j < s) red[j] += red[j + s];
    __syncthreads();
  }
  probs[(size_t)row * C + j] = (__bf16)(ex / red[0]);
}

// ---------------- kernel 4: context[r][i][h][d] = sum_j probs[h][i][j]*v[h][r][d][j] ----------------
// One block (8 waves) per (h, r): waves tile the full 256x64 output as 4x2 macro-tiles.
__global__ void context_kernel(const __bf16* __restrict__ probs,
                               const __bf16* __restrict__ vT,
                               __bf16* __restrict__ ctx) {
  const int lane16 = threadIdx.x & 15;
  const int half   = (threadIdx.x >> 4) & 1;
  const int w      = threadIdx.x >> 5;   // 0..7
  const int hr     = blockIdx.x;         // h*R + r
  const int h      = hr / R;
  const int r      = hr % R;
  const int m0     = (w >> 1) * 64;      // 4 wave-rows over i (256)
  const int n0     = (w & 1) * 32;       // 2 wave-cols over d (64)

  const __bf16* A = probs + (size_t)h * C * C;            // row i, K=j
  const __bf16* B = vT + ((size_t)h * R + r) * D * C;     // row d, K=j

  v8f acc[4][2] = {};
  gemm_core_4x2(A, (size_t)C, B, (size_t)C, m0, n0, C, acc);

#pragma unroll
  for (int mt = 0; mt < 4; ++mt)
#pragma unroll
    for (int nt = 0; nt < 2; ++nt)
#pragma unroll
      for (int e = 0; e < 8; ++e) {
        int i = m0 + mt * 16 + e + 8 * half;
        int d = n0 + nt * 16 + lane16;
        ctx[((size_t)r * C + i) * E + h * D + d] = (__bf16)acc[mt][nt][e];
      }
}

// ---------------- kernel 5: output projection, fp32 out ----------------
__global__ void out_kernel(const __bf16* __restrict__ ctx,
                           const __bf16* __restrict__ woT,
                           const float* __restrict__ bo,
                           float* __restrict__ out) {
  const int lane16 = threadIdx.x & 15;
  const int half   = (threadIdx.x >> 4) & 1;
  const int wave   = blockIdx.x * (blockDim.x >> 5) + (threadIdx.x >> 5);
  const int nw_n   = E / 32;
  const int m0     = (wave / nw_n) * 64;
  const int n0     = (wave % nw_n) * 32;

  v8f acc[4][2] = {};
  gemm_core_4x2(ctx, (size_t)E, woT, (size_t)E, m0, n0, E, acc);

#pragma unroll
  for (int mt = 0; mt < 4; ++mt)
#pragma unroll
    for (int nt = 0; nt < 2; ++nt)
#pragma unroll
      for (int e = 0; e < 8; ++e) {
        int mi = m0 + mt * 16 + e + 8 * half;
        int ni = n0 + nt * 16 + lane16;
        out[(size_t)mi * E + ni] = acc[mt][nt][e] + bo[ni];
      }
}

// ---------------- host: workspace carving + launches ----------------
extern "C" void kernel_launch(void* const* d_in, const int* in_sizes, int n_in,
                              void* d_out, int out_size, void* d_ws, size_t ws_size,
                              hipStream_t stream) {
  const float* x  = (const float*)d_in[0];
  const float* wq = (const float*)d_in[1];
  const float* bq = (const float*)d_in[2];
  const float* wk = (const float*)d_in[3];
  const float* bk = (const float*)d_in[4];
  const float* wv = (const float*)d_in[5];
  const float* bv = (const float*)d_in[6];
  const float* wo = (const float*)d_in[7];
  const float* bo = (const float*)d_in[8];
  float* out = (float*)d_out;

  char* ws = (char*)d_ws;
  size_t off = 0;
  auto carve = [&](size_t bytes) {
    void* p = ws + off;
    off = (off + bytes + 255) & ~(size_t)255;
    return p;
  };
  const size_t wT_bytes   = (size_t)E * E * sizeof(__bf16);          // 1.18 MB
  const size_t qkv_bytes  = (size_t)MTOK * E * sizeof(__bf16);       // 50.3 MB
  const size_t log_bytes  = (size_t)H * C * C * sizeof(float);       // 3.1 MB
  const size_t prob_bytes = (size_t)H * C * C * sizeof(__bf16);      // 1.6 MB

  __bf16* wqkvT = (__bf16*)carve(3 * wT_bytes);  // [3E][E] : q rows, k rows, v rows
  __bf16* wTo   = (__bf16*)carve(wT_bytes);
  __bf16* xb    = (__bf16*)carve(qkv_bytes);     // x in bf16
  __bf16* q_t   = (__bf16*)carve(qkv_bytes);     // [H][C][R*D]
  __bf16* k_t   = (__bf16*)carve(qkv_bytes);     // [H][C][R*D]
  __bf16* v_t   = (__bf16*)carve(qkv_bytes);     // [H][R][D][C]
  float*  lgts  = (float*) carve(log_bytes);     // [H][C][C]
  __bf16* probs = (__bf16*)carve(prob_bytes);    // [H][C][C]
  __bf16* ctx   = (__bf16*)carve(qkv_bytes);     // [MTOK][E]
  (void)ws_size; (void)n_in; (void)in_sizes; (void)out_size;

  const int TPB = 256;                  // 8 waves per block
  const int WPB = TPB / 32;

  // 0) x -> bf16 ; weight transposes to bf16 (N x K), QKV concatenated
  xcvt_kernel<<<(MTOK * E) / (TPB * 4), TPB, 0, stream>>>(x, xb);
  const int tblocks = (E * E) / TPB;    // 2304
  wtrans_kernel<<<tblocks, TPB, 0, stream>>>(wq, wqkvT + 0 * (size_t)E * E);
  wtrans_kernel<<<tblocks, TPB, 0, stream>>>(wk, wqkvT + 1 * (size_t)E * E);
  wtrans_kernel<<<tblocks, TPB, 0, stream>>>(wv, wqkvT + 2 * (size_t)E * E);
  wtrans_kernel<<<tblocks, TPB, 0, stream>>>(wo, wTo);

  // 1) fused QKV projection: (32768/64) x (2304/32) = 512 x 72 = 36864 waves
  const float q_scale = (1.0f / 8.0f) / sqrtf((float)R);   // D^-0.5 / sqrt(R)
  const int qkv_blocks = ((MTOK / 64) * (NQKV / 32)) / WPB;  // 4608
  qkv_kernel<<<qkv_blocks, TPB, 0, stream>>>(xb, wqkvT, bq, bk, bv,
                                             q_t, k_t, v_t, q_scale);

  // 2) tied logits: 12 heads * (4x8) waves = 384 waves
  logits_kernel<<<(H * 32) / WPB, TPB, 0, stream>>>(q_t, k_t, lgts);

  // 3) softmax: one block per (h, i) row
  softmax_kernel<<<H * C, C, 0, stream>>>(lgts, probs);

  // 4) context: one block per (h, r)
  context_kernel<<<H * R, TPB, 0, stream>>>(probs, v_t, ctx);

  // 5) output projection
  out_kernel<<<((MTOK / 64) * (E / 32)) / WPB, TPB, 0, stream>>>(ctx, wTo, bo, out);
}
